// MultiHeadMALAAttention_55997783605382
// MI455X (gfx1250) — matrix operand order repaired and checked by reading, not compiled, not run on hardware
//
#include <hip/hip_runtime.h>
#include <hip/hip_bf16.h>

// ---- Problem constants --------------------------------------------------
#define BATCH     4
#define SEQ       8192
#define DIM       256
#define NHEAD     8
#define HDIM      32
#define INTERNAL  256            // NHEAD * HDIM
#define QKVO_COLS 1024           // 4 * INTERNAL
#define MROWS     (BATCH * SEQ)  // 32768
#define SCALE_F   0.17677669529663687f   // 32^-0.5
#define STATS_PER_BH 1088        // 32*32 kv + 32 ksum + 32 vsum

typedef __attribute__((ext_vector_type(16))) __bf16 v16bf;
typedef __attribute__((ext_vector_type(8)))  __bf16 bf8;
typedef __attribute__((ext_vector_type(8)))  float  v8f;
typedef __attribute__((ext_vector_type(4)))  unsigned int u32x4;
typedef __attribute__((ext_vector_type(8)))  unsigned int u32x8;

// ---- WMMA fragment helpers (layouts per CDNA5 ISA 7.12.2, wave32) -------
// A 16x32 bf16: lane l -> row m=l&15; K runs [kg..kg+7] and [kg+16..kg+23],
// kg = (l>=16 ? 8 : 0).
__device__ __forceinline__ v16bf load_a_frag_f32(const float* __restrict__ A,
                                                 int lda, int lane) {
  int m  = lane & 15;
  int kg = (lane >> 4) << 3;
  const float* p0 = A + m * lda + kg;
  const float* p1 = p0 + 16;
  v16bf f;
#pragma unroll
  for (int j = 0; j < 8; ++j) { f[j] = (__bf16)p0[j]; f[j + 8] = (__bf16)p1[j]; }
  return f;
}

// B 32x16 bf16 from row-major weights (out col n <- weight row n, K contig):
// lane l -> col n=l&15; K runs [kg..kg+7] and [kg+16..kg+23], kg=(l>=16?8:0).
__device__ __forceinline__ v16bf load_b_frag_f32(const float* __restrict__ W,
                                                 int ldb, int lane) {
  int n  = lane & 15;
  int ks = (lane >> 4) << 3;
  const float* p = W + n * ldb + ks;
  v16bf f;
#pragma unroll
  for (int j = 0; j < 8; ++j) { f[j] = (__bf16)p[j]; f[j + 8] = (__bf16)p[j + 16]; }
  return f;
}

// ---- TDM: async 2D f32 tile load, Global -> padded LDS ------------------
// D# per CDNA5 ISA §8.3/§8.4.  2-group form (tensors up to 2D), data_size=4B.
// pad_enable: interval = 32 DWORDs (code 4), amount = 4 DWORDs (code 3)
// => hardware writes LDS rows with stride 36 floats (bank-conflict free).
__device__ __forceinline__ void tdm_load_tile32x32_f32(unsigned int lds_off,
                                                       const float* gptr,
                                                       unsigned int row_stride_elems) {
  unsigned long long ga = (unsigned long long)(const void*)gptr;
  u32x4 g0;
  g0[0] = 1u;                                   // count=1 (valid user descriptor)
  g0[1] = lds_off;                              // lds_addr (bytes)
  g0[2] = (unsigned int)ga;                     // global_addr[31:0]
  g0[3] = (unsigned int)(ga >> 32) | (2u << 30);// global_addr[56:32] | type=2
  u32x8 g1;
  g1[0] = (2u << 16)      // data_size = 4B
        | (1u << 20)      // pad_enable
        | (4u << 22)      // pad_interval: 32 DWORDs
        | (3u << 25);     // pad_amount:   4 DWORDs
  g1[1] = (32u & 0xFFFFu) << 16;                // tensor_dim0[15:0] (=32)
  g1[2] = (32u >> 16) | ((32u & 0xFFFFu) << 16);// tensor_dim0 hi | tensor_dim1 lo
  g1[3] = (32u >> 16) | (32u << 16);            // tensor_dim1 hi | tile_dim0=32
  g1[4] = 32u;                                  // tile_dim1=32, tile_dim2=0
  g1[5] = row_stride_elems;                     // tensor_dim0_stride[31:0]
  g1[6] = 0u;                                   // stride hi / dim1_stride lo
  g1[7] = 0u;
  asm volatile("tensor_load_to_lds %0, %1" :: "s"(g0), "s"(g1) : "memory");
}

// ---- Kernel 0: zero the stats accumulator -------------------------------
__global__ __launch_bounds__(256) void k_zero(float* __restrict__ p, int n) {
  int i = blockIdx.x * 256 + threadIdx.x;
  if (i < n) p[i] = 0.0f;
}

// ---- Kernel 1: qkvo = x @ W_qkvo^T + b  (M=32768, N=1024, K=256) --------
// 8 waves/block; each wave owns a 16x64 strip (4 accumulators, A reused 4x).
__global__ __launch_bounds__(256) void k_qkvo(const float* __restrict__ x,
                                              const float* __restrict__ W,
                                              const float* __restrict__ bias,
                                              float* __restrict__ qkvo) {
  int lane  = threadIdx.x & 31;
  int wave  = threadIdx.x >> 5;
  int mtile = blockIdx.x >> 1;
  int nb0   = (blockIdx.x & 1) * 32 + wave * 4;   // first of 4 N-tiles

  const float* Abase = x + (size_t)(mtile * 16) * DIM;
  const float* B0 = W + (size_t)((nb0 + 0) * 16) * DIM;
  const float* B1 = W + (size_t)((nb0 + 1) * 16) * DIM;
  const float* B2 = W + (size_t)((nb0 + 2) * 16) * DIM;
  const float* B3 = W + (size_t)((nb0 + 3) * 16) * DIM;

  v8f acc0 = {}, acc1 = {}, acc2 = {}, acc3 = {};
#pragma unroll
  for (int kk = 0; kk < DIM; kk += 32) {
    __builtin_prefetch(Abase + kk + 64, 0, 1);   // -> global_prefetch_b8
    v16bf a  = load_a_frag_f32(Abase + kk, DIM, lane);
    v16bf b0 = load_b_frag_f32(B0 + kk, DIM, lane);
    v16bf b1 = load_b_frag_f32(B1 + kk, DIM, lane);
    v16bf b2 = load_b_frag_f32(B2 + kk, DIM, lane);
    v16bf b3 = load_b_frag_f32(B3 + kk, DIM, lane);
    acc0 = __builtin_amdgcn_wmma_f32_16x16x32_bf16(false, a, false, b0, (short)0, acc0, false, false);
    acc1 = __builtin_amdgcn_wmma_f32_16x16x32_bf16(false, a, false, b1, (short)0, acc1, false, false);
    acc2 = __builtin_amdgcn_wmma_f32_16x16x32_bf16(false, a, false, b2, (short)0, acc2, false, false);
    acc3 = __builtin_amdgcn_wmma_f32_16x16x32_bf16(false, a, false, b3, (short)0, acc3, false, false);
  }
  int n  = lane & 15;
  int mb = (lane >> 4) << 3;
  v8f accs[4] = {acc0, acc1, acc2, acc3};
#pragma unroll
  for (int j = 0; j < 4; ++j) {
    float bv = bias[(nb0 + j) * 16 + n];
    float* C = qkvo + (size_t)(mtile * 16) * QKVO_COLS + (nb0 + j) * 16;
#pragma unroll
    for (int r = 0; r < 8; ++r)
      C[(size_t)(mb + r) * QKVO_COLS + n] = accs[j][r] + bv;
  }
}

// ---- Kernel 2: per-(b,h) reductions: kv_state, ksum, vsum ---------------
// grid = 32 (b,h) * 16 N-chunks; TDM stages raw k/v tiles into padded LDS,
// VALU does elu/theta_shift/outer-product; atomicAdd f32 partials.
#define NCHUNK 16
#define CHTOK  (SEQ / NCHUNK)   // 512
__global__ __launch_bounds__(256) void k_stats(const float* __restrict__ qkvo,
                                               const float* __restrict__ sinT,
                                               const float* __restrict__ cosT,
                                               float* __restrict__ stats) {
  __shared__ __align__(16) float s_kraw[32][36];   // TDM destination (k)
  __shared__ __align__(16) float s_v   [32][36];   // TDM destination (v)
  __shared__ __align__(16) float s_kh  [32][36];
  __shared__ __align__(16) float s_ks  [32][36];

  int tid   = threadIdx.x;
  int bh    = blockIdx.x & 31;
  int chunk = blockIdx.x >> 5;
  int b = bh >> 3, h = bh & 7;
  int nbase = chunk * CHTOK;

  int dd = tid >> 3;            // 0..31
  int e0 = (tid & 7) << 2;      // 0,4,...,28
  float a0 = 0.f, a1 = 0.f, a2 = 0.f, a3 = 0.f;
  float ksum = 0.f, vsum = 0.f;

  unsigned int lds_k = (unsigned int)(unsigned long long)&s_kraw[0][0];
  unsigned int lds_v = (unsigned int)(unsigned long long)&s_v[0][0];

  for (int tt = 0; tt < CHTOK; tt += 32) {
    int n0 = nbase + tt;
    size_t row0 = (size_t)(b * SEQ + n0) * QKVO_COLS;
    if (tid < 32) {   // wave 0 issues both tensor DMAs
      tdm_load_tile32x32_f32(lds_k, qkvo + row0 + INTERNAL     + h * HDIM, QKVO_COLS);
      tdm_load_tile32x32_f32(lds_v, qkvo + row0 + 2 * INTERNAL + h * HDIM, QKVO_COLS);
      __builtin_amdgcn_s_wait_tensorcnt(0);
    }
    __syncthreads();

    for (int ei = tid; ei < 1024; ei += 256) {
      int t = ei >> 5, d = ei & 31;
      float kraw = s_kraw[t][d];
      s_kh[t][d] = kraw > 0.f ? kraw + 1.f : __expf(kraw);  // elu+1
    }
    __syncthreads();
    for (int ei = tid; ei < 1024; ei += 256) {
      int t = ei >> 5, d = ei & 31;
      int n = n0 + t;
      float kh  = s_kh[t][d];
      float par = s_kh[t][d ^ 1];
      float rot = (d & 1) ? par : -par;       // rotate_every_two
      s_ks[t][d] = kh * cosT[n * HDIM + d] + rot * sinT[n * HDIM + d];
    }
    __syncthreads();
#pragma unroll 4
    for (int t = 0; t < 32; ++t) {
      float kd = s_ks[t][dd];
      float4 vv = *(const float4*)&s_v[t][e0];
      a0 += kd * vv.x; a1 += kd * vv.y; a2 += kd * vv.z; a3 += kd * vv.w;
    }
    if (tid < 32) {
#pragma unroll 8
      for (int t = 0; t < 32; ++t) { ksum += s_kh[t][tid]; vsum += s_v[t][tid]; }
    }
    __syncthreads();   // protect s_kraw/s_v before next TDM overwrite
  }

  const float s2 = SCALE_F / (float)SEQ;      // s^2 folded into kv_state
  float* st = stats + (size_t)bh * STATS_PER_BH;
  atomicAdd(&st[dd * 32 + e0 + 0], a0 * s2);
  atomicAdd(&st[dd * 32 + e0 + 1], a1 * s2);
  atomicAdd(&st[dd * 32 + e0 + 2], a2 * s2);
  atomicAdd(&st[dd * 32 + e0 + 3], a3 * s2);
  if (tid < 32) {
    atomicAdd(&st[1024 + tid], ksum);
    atomicAdd(&st[1056 + tid], vsum);
  }
}

// ---- Kernel 3: fused epilogue + out = u @ W_proj^T + b ------------------
// One block = 64 tokens of one batch. Phase1: compute gated activation u
// (elu, z, theta_shift, qs@kv, lepe conv3, *o) -> bf16 LDS (stride 264).
// Phase2: 8 waves x 8 WMMA tiles vs W_proj, fused bias + f32 store.
#define ULD 264                 // padded bf16 row stride (16B aligned, conflict-free)
__global__ __launch_bounds__(256) void k_fused(const float* __restrict__ qkvo,
                                               const float* __restrict__ sinT,
                                               const float* __restrict__ cosT,
                                               const float* __restrict__ stats,
                                               const float* __restrict__ Wlepe,
                                               const float* __restrict__ blepe,
                                               const float* __restrict__ Wproj,
                                               const float* __restrict__ bproj,
                                               float* __restrict__ out) {
  extern __shared__ __align__(16) char smem[];
  float*  s_kv = (float*)smem;                       // 8*1024 f32
  float*  s_km = s_kv + NHEAD * 1024;                // 8*32
  float*  s_vm = s_km + NHEAD * HDIM;                // 8*32
  __bf16* s_u  = (__bf16*)(s_vm + NHEAD * HDIM);     // 64*ULD bf16

  int tid = threadIdx.x, lane = tid & 31, wave = tid >> 5;
  int b  = blockIdx.x >> 7;                // 128 blocks per batch
  int t0 = (blockIdx.x & 127) * 64;

  // phase 0: stats -> LDS (means normalized here)
  for (int i = tid; i < NHEAD * STATS_PER_BH; i += 256) {
    int h = i / STATS_PER_BH, r = i % STATS_PER_BH;
    float v = stats[(size_t)(b * NHEAD + h) * STATS_PER_BH + r];
    if (r < 1024)      s_kv[h * 1024 + r] = v;
    else if (r < 1056) s_km[h * HDIM + (r - 1024)] = v * (1.0f / (float)SEQ);
    else               s_vm[h * HDIM + (r - 1056)] = v * (1.0f / (float)SEQ);
  }
  __syncthreads();

  // phase 1: two (token, head) pairs per thread
#pragma unroll 1
  for (int pp = 0; pp < 2; ++pp) {
    int p = tid * 2 + pp;
    int t = p >> 3, h = p & 7;
    int n = t0 + t;
    size_t row = (size_t)(b * SEQ + n) * QKVO_COLS;
    const float* qp = qkvo + row + h * HDIM;
    const float* vp = qkvo + row + 2 * INTERNAL + h * HDIM;
    const float* op = qkvo + row + 3 * INTERNAL + h * HDIM;

    float qh[HDIM];
    float z = 0.f;
#pragma unroll
    for (int d = 0; d < HDIM; ++d) {
      float q = qp[d];
      float e = q > 0.f ? q + 1.f : __expf(q);
      qh[d] = e;
      z += e * s_km[h * HDIM + d];
    }
    z *= SCALE_F;

    float qs[HDIM];
#pragma unroll
    for (int d = 0; d < HDIM; d += 2) {
      float c0 = cosT[n * HDIM + d], c1 = cosT[n * HDIM + d + 1];
      float s0 = sinT[n * HDIM + d], s1 = sinT[n * HDIM + d + 1];
      qs[d]     = qh[d]     * c0 - qh[d + 1] * s0;
      qs[d + 1] = qh[d + 1] * c1 + qh[d]     * s1;
    }

    float gain = 1.0f + 1.0f / (z + 1e-6f);
#pragma unroll 1
    for (int e = 0; e < HDIM; ++e) {
      float a = 0.f;
#pragma unroll
      for (int d = 0; d < HDIM; ++d) a += qs[d] * s_kv[h * 1024 + d * 32 + e];
      float res = a * gain - z * s_vm[h * HDIM + e];

      int c = h * HDIM + e;
      float lep = blepe[c];
      float w0 = Wlepe[c * 3 + 0], w1 = Wlepe[c * 3 + 1], w2 = Wlepe[c * 3 + 2];
      float vm1 = (n > 0)       ? qkvo[row - QKVO_COLS + 2 * INTERNAL + c] : 0.f;
      float vp1 = (n < SEQ - 1) ? qkvo[row + QKVO_COLS + 2 * INTERNAL + c] : 0.f;
      lep += vm1 * w0 + vp[e] * w1 + vp1 * w2;

      float u = (res + lep) * op[e];
      s_u[t * ULD + c] = (__bf16)u;
    }
  }
  __syncthreads();

  // phase 2: out(64x256) = u(64x256) @ Wproj^T(256x256)
#pragma unroll 1
  for (int i = 0; i < 8; ++i) {
    int tt = wave * 8 + i;
    int mt = tt >> 4, nt = tt & 15;
    const float* Bbase = Wproj + (size_t)(nt * 16) * INTERNAL;
    int mrow = mt * 16 + (lane & 15);
    int kg   = (lane >> 4) << 3;

    v8f acc = {};
#pragma unroll
    for (int kk = 0; kk < INTERNAL; kk += 32) {
      bf8 a0 = *(const bf8*)&s_u[mrow * ULD + kk + kg];        // ds_load_b128
      bf8 a1 = *(const bf8*)&s_u[mrow * ULD + kk + kg + 16];
      v16bf a;
#pragma unroll
      for (int j = 0; j < 8; ++j) { a[j] = a0[j]; a[j + 8] = a1[j]; }
      v16bf bfr = load_b_frag_f32(Bbase + kk, INTERNAL, lane);
      acc = __builtin_amdgcn_wmma_f32_16x16x32_bf16(false, a, false, bfr,
                                                    (short)0, acc, false, false);
    }
    int nn = lane & 15;
    int mb = (lane >> 4) << 3;
    float bv = bproj[nt * 16 + nn];
    float* C = out + (size_t)(b * SEQ + t0 + mt * 16) * DIM + nt * 16;
#pragma unroll
    for (int r = 0; r < 8; ++r) C[(size_t)(mb + r) * DIM + nn] = acc[r] + bv;
  }
}

// ---- Host-side launch ---------------------------------------------------
extern "C" void kernel_launch(void* const* d_in, const int* in_sizes, int n_in,
                              void* d_out, int out_size, void* d_ws, size_t ws_size,
                              hipStream_t stream) {
  const float* x     = (const float*)d_in[0];
  const float* sinT  = (const float*)d_in[1];
  const float* cosT  = (const float*)d_in[2];
  const float* Wqkvo = (const float*)d_in[3];
  const float* bqkvo = (const float*)d_in[4];
  const float* Wlepe = (const float*)d_in[5];
  const float* blepe = (const float*)d_in[6];
  const float* Wproj = (const float*)d_in[7];
  const float* bproj = (const float*)d_in[8];
  float* out = (float*)d_out;

  float* qkvo  = (float*)d_ws;                          // 32768*1024 f32
  float* stats = qkvo + (size_t)MROWS * QKVO_COLS;      // 32*1088 f32

  const int statsN = 32 * STATS_PER_BH;
  k_zero<<<(statsN + 255) / 256, 256, 0, stream>>>(stats, statsN);

  // GEMM1: 2048 M-tiles * 2 half-row blocks (each wave: 16x64 strip)
  k_qkvo<<<(MROWS / 16) * 2, 256, 0, stream>>>(x, Wqkvo, bqkvo, qkvo);

  // reductions: 32 (b,h) * 16 chunks (TDM-staged tiles)
  k_stats<<<32 * NCHUNK, 256, 0, stream>>>(qkvo, sinT, cosT, stats);

  // fused epilogue + proj GEMM: 512 blocks of 64 tokens
  size_t smem = (size_t)(NHEAD * 1024 + 2 * NHEAD * HDIM) * sizeof(float)
              + (size_t)64 * ULD * sizeof(__bf16);
  k_fused<<<(MROWS / 64), 256, smem, stream>>>(qkvo, sinT, cosT, stats,
                                               Wlepe, blepe, Wproj, bproj, out);
}
